// ScaledProductAttention_87857851007383
// MI455X (gfx1250) — matrix operand order repaired and checked by reading, not compiled, not run on hardware
//
#include <hip/hip_runtime.h>
#include <hip/hip_bf16.h>

// ---------------------------------------------------------------------------
// CDNA5 (gfx1250, wave32) attention pipeline:
//   1) cast weights f32 -> f16
//   2) pq/pk/pv = X @ W^T + b          (WMMA f16, f32 accum, f16 out)
//   3) scores   = pq @ pk^T  (batched) (WMMA f16, f32 out)
//   4) P        = softmax(mask ? scores/8 : -inf)  -> f16
//   5) mh       = P @ pv     (batched) (WMMA f16, f16 out)
//   6) out      = mh @ Wo^T + bo       (WMMA f16, f32 out)
// All matmul dims are multiples of 128/32 -> no bounds checks.
// ---------------------------------------------------------------------------

typedef __attribute__((ext_vector_type(16))) _Float16 v16h;
typedef __attribute__((ext_vector_type(8)))  _Float16 v8h;
typedef __attribute__((ext_vector_type(4)))  _Float16 v4h;
typedef __attribute__((ext_vector_type(8)))  float    v8f;
typedef __attribute__((ext_vector_type(4)))  float    v4f;

#define LDS_PITCH 40   // halfs per row: 32 data + 8 pad (80 B, 16 B aligned, conflict-free)

// A fragment (16x32 f16, rows m): lane l holds row m=l&15.
// half h -> k = (h&7) + ((h>=8)?16:0) + ((l>=16)?8:0)   [ISA 7.12.2]
__device__ __forceinline__ v16h load_frag_a(const _Float16* s, int rowBase, int lane) {
  const int r = lane & 15, hi = lane >> 4;
  const _Float16* p = s + (rowBase + r) * LDS_PITCH + hi * 8;
  union { v16h v; v8h h[2]; } u;
  u.h[0] = *(const v8h*)(p);
  u.h[1] = *(const v8h*)(p + 16);
  return u.v;
}

// B fragment (32x16 f16, cols n, B staged as [n][k] in LDS): lane l holds col n=l&15.
// half h -> k = (h&7) + ((h>=8)?16:0) + ((l>=16)?8:0)   (same k order as A)
__device__ __forceinline__ v16h load_frag_b(const _Float16* s, int colBase, int lane) {
  const int n = lane & 15, hi = lane >> 4;
  const _Float16* p = s + (colBase + n) * LDS_PITCH + hi * 8;
  union { v16h v; v8h h[2]; } u;
  u.h[0] = *(const v8h*)(p);
  u.h[1] = *(const v8h*)(p + 16);
  return u.v;
}

// C = A @ op(B) (+bias). A: [M,K] row-major (f32 or f16).
// B_IS_KN=false: B is [N,K] row-major f16 (i.e. C = A @ B^T with B^T implicit: weights, pk)
// B_IS_KN=true : B is [K,N] row-major f16 (C = A @ B), staged transposed into LDS.
// CT: _Float16 or float output.
template <typename AT, bool B_IS_KN, bool BIAS, typename CT>
__global__ __launch_bounds__(256)
void gemm128x128(const AT* __restrict__ A, const _Float16* __restrict__ B,
                 const float* __restrict__ bias, CT* __restrict__ C,
                 int M, int N, int K,
                 long long aBatch, long long bBatch, long long cBatch)
{
  __shared__ _Float16 As[128 * LDS_PITCH];
  __shared__ _Float16 Bs[128 * LDS_PITCH];

  A += (long long)blockIdx.z * aBatch;
  B += (long long)blockIdx.z * bBatch;
  C += (long long)blockIdx.z * cBatch;

  const int m0   = blockIdx.y * 128;
  const int n0   = blockIdx.x * 128;
  const int tid  = threadIdx.x;
  const int lane = tid & 31;
  const int wave = tid >> 5;
  const int wm   = (wave & 3) * 32;   // wave's m offset inside tile (4 waves down M)
  const int wn   = (wave >> 2) * 64;  // wave's n offset inside tile (2 waves across N)

  v8f acc[2][4];
  #pragma unroll
  for (int i = 0; i < 2; ++i)
    #pragma unroll
    for (int j = 0; j < 4; ++j)
      #pragma unroll
      for (int e = 0; e < 8; ++e) acc[i][j][e] = 0.0f;

  const int sRow = tid >> 1;        // 0..127: staging row (A and NT-B)
  const int sK   = (tid & 1) * 16;  // half-row split
  const int tK   = tid >> 3;        // 0..31 : staging k (NN-B)
  const int tN   = (tid & 7) * 16;  // 0..112: staging n (NN-B)

  for (int k0 = 0; k0 < K; k0 += 32) {
    // ---- stage A tile [128 x 32] ----
    {
      const AT* ga = A + (long long)(m0 + sRow) * K + (k0 + sK);
      if (k0 + 32 < K) __builtin_prefetch(ga + 32, 0, 1);  // global_prefetch_b8
      v8h h0, h1;
      if constexpr (sizeof(AT) == 4) {            // f32 source: convert in flight
        const v4f* gv = (const v4f*)ga;
        v4f f0 = gv[0], f1 = gv[1], f2 = gv[2], f3 = gv[3];
        #pragma unroll
        for (int i = 0; i < 4; ++i) {
          h0[i] = (_Float16)f0[i]; h0[4 + i] = (_Float16)f1[i];
          h1[i] = (_Float16)f2[i]; h1[4 + i] = (_Float16)f3[i];
        }
      } else {
        const v8h* gv = (const v8h*)ga;
        h0 = gv[0]; h1 = gv[1];
      }
      *(v8h*)(&As[sRow * LDS_PITCH + sK])     = h0;
      *(v8h*)(&As[sRow * LDS_PITCH + sK + 8]) = h1;
    }
    // ---- stage B tile as [n][k] in LDS ----
    if constexpr (!B_IS_KN) {   // B is [N,K]: direct copy
      const _Float16* gb = B + (long long)(n0 + sRow) * K + (k0 + sK);
      if (k0 + 32 < K) __builtin_prefetch(gb + 32, 0, 1);
      v8h h0 = ((const v8h*)gb)[0];
      v8h h1 = ((const v8h*)gb)[1];
      *(v8h*)(&Bs[sRow * LDS_PITCH + sK])     = h0;
      *(v8h*)(&Bs[sRow * LDS_PITCH + sK + 8]) = h1;
    } else {                    // B is [K,N]: transpose while staging
      const _Float16* gb = B + (long long)(k0 + tK) * N + (n0 + tN);
      if (k0 + 32 < K) __builtin_prefetch(gb + 32LL * N, 0, 1);
      v8h h0 = ((const v8h*)gb)[0];
      v8h h1 = ((const v8h*)gb)[1];
      #pragma unroll
      for (int j = 0; j < 8; ++j) {
        Bs[(tN + j) * LDS_PITCH + tK]     = h0[j];
        Bs[(tN + 8 + j) * LDS_PITCH + tK] = h1[j];
      }
    }
    __syncthreads();

    v16h afrag[2], bfrag[4];
    #pragma unroll
    for (int i = 0; i < 2; ++i) afrag[i] = load_frag_a(As, wm + i * 16, lane);
    #pragma unroll
    for (int j = 0; j < 4; ++j) bfrag[j] = load_frag_b(Bs, wn + j * 16, lane);

    #pragma unroll
    for (int i = 0; i < 2; ++i)
      #pragma unroll
      for (int j = 0; j < 4; ++j)
        acc[i][j] = __builtin_amdgcn_wmma_f32_16x16x32_f16(
            /*neg_a=*/false, afrag[i], /*neg_b=*/false, bfrag[j],
            /*c_mod=*/(short)0, acc[i][j], /*reuse_a=*/false, /*reuse_b=*/false);

    __syncthreads();
  }

  // ---- epilogue: C element (e, lane) -> row = base + e + 8*(lane>=16), col = lane&15 ----
  const int r = lane & 15, hi = lane >> 4;
  #pragma unroll
  for (int j = 0; j < 4; ++j) {
    const int col = n0 + wn + j * 16 + r;
    float bv = 0.0f;
    if constexpr (BIAS) bv = bias[col];
    #pragma unroll
    for (int i = 0; i < 2; ++i) {
      const int mBase = m0 + wm + i * 16 + hi * 8;
      #pragma unroll
      for (int e = 0; e < 8; ++e) {
        const float v = acc[i][j][e] + bv;
        C[(long long)(mBase + e) * N + col] = (CT)v;
      }
    }
  }
}

// Masked, scaled softmax over one row of 2048. val = mask ? s/8 : -inf.
__global__ __launch_bounds__(256)
void softmax_mask_kernel(const float* __restrict__ S, const int* __restrict__ mask,
                         _Float16* __restrict__ P, int SK)
{
  const long long row = blockIdx.x;
  const float* s = S + row * SK;
  const int*   m = mask + row * SK;
  _Float16*    p = P + row * SK;
  const int tid = threadIdx.x;
  __shared__ float red[256];

  const float NEG_INF = -__builtin_inff();
  float vals[8];
  float mx = NEG_INF;
  #pragma unroll
  for (int i = 0; i < 8; ++i) {
    const int c = tid + i * 256;
    const float v = (m[c] == 0) ? NEG_INF : s[c] * 0.125f;  // 1/sqrt(64)
    vals[i] = v;
    mx = fmaxf(mx, v);
  }
  red[tid] = mx; __syncthreads();
  for (int o = 128; o > 0; o >>= 1) {
    if (tid < o) red[tid] = fmaxf(red[tid], red[tid + o]);
    __syncthreads();
  }
  mx = red[0]; __syncthreads();

  float sum = 0.0f;
  #pragma unroll
  for (int i = 0; i < 8; ++i) {
    const float e = (vals[i] == NEG_INF) ? 0.0f : __expf(vals[i] - mx);
    vals[i] = e; sum += e;
  }
  red[tid] = sum; __syncthreads();
  for (int o = 128; o > 0; o >>= 1) {
    if (tid < o) red[tid] += red[tid + o];
    __syncthreads();
  }
  const float inv = 1.0f / red[0];  // all-masked row -> 0*inf = NaN, matches reference
  #pragma unroll
  for (int i = 0; i < 8; ++i) p[tid + i * 256] = (_Float16)(vals[i] * inv);
}

__global__ __launch_bounds__(256)
void cast_f32_to_f16_kernel(const float* __restrict__ src, _Float16* __restrict__ dst, int n)
{
  const int i = (blockIdx.x * 256 + threadIdx.x) * 4;
  if (i + 3 < n) {
    const v4f f = *(const v4f*)(src + i);
    v4h h;
    #pragma unroll
    for (int k = 0; k < 4; ++k) h[k] = (_Float16)f[k];
    *(v4h*)(dst + i) = h;
  }
}

extern "C" void kernel_launch(void* const* d_in, const int* in_sizes, int n_in,
                              void* d_out, int out_size, void* d_ws, size_t ws_size,
                              hipStream_t stream)
{
  (void)in_sizes; (void)n_in; (void)out_size; (void)ws_size;
  const int B = 4, SQ = 2048, SK = 2048, H = 1024;
  const int MS = B * SQ;  // 8192 flattened rows

  const float* Q    = (const float*)d_in[0];
  const float* Kin  = (const float*)d_in[1];
  const float* V    = (const float*)d_in[2];
  const int*   mask = (const int*)d_in[3];
  const float* Wq = (const float*)d_in[4];  const float* bq = (const float*)d_in[5];
  const float* Wk = (const float*)d_in[6];  const float* bk = (const float*)d_in[7];
  const float* Wv = (const float*)d_in[8];  const float* bv = (const float*)d_in[9];
  const float* Wo = (const float*)d_in[10]; const float* bo = (const float*)d_in[11];
  float* out = (float*)d_out;

  // Workspace layout (120 MB total, with temporal reuse):
  char* ws = (char*)d_ws;
  const size_t MB = 1024ull * 1024ull;
  _Float16* wq_h = (_Float16*)(ws + 0 * MB);   // 2 MB
  _Float16* wk_h = (_Float16*)(ws + 2 * MB);   // 2 MB
  _Float16* wv_h = (_Float16*)(ws + 4 * MB);   // 2 MB
  _Float16* wo_h = (_Float16*)(ws + 6 * MB);   // 2 MB
  _Float16* pv   = (_Float16*)(ws + 8 * MB);   // 16 MB  (live until step 5)
  _Float16* pq   = (_Float16*)(ws + 24 * MB);  // 16 MB  (dead after step 3)
  _Float16* pk   = (_Float16*)(ws + 40 * MB);  // 16 MB  (dead after step 3)
  float*    sc   = (float*)   (ws + 56 * MB);  // 64 MB  (dead after step 4)
  _Float16* P    = (_Float16*)(ws + 24 * MB);  // 32 MB  overlays pq+pk
  _Float16* mh   = (_Float16*)(ws + 56 * MB);  // 16 MB  overlays sc

  // 1) cast weights to f16
  cast_f32_to_f16_kernel<<<dim3(H * H / 1024), dim3(256), 0, stream>>>(Wq, wq_h, H * H);
  cast_f32_to_f16_kernel<<<dim3(H * H / 1024), dim3(256), 0, stream>>>(Wk, wk_h, H * H);
  cast_f32_to_f16_kernel<<<dim3(H * H / 1024), dim3(256), 0, stream>>>(Wv, wv_h, H * H);
  cast_f32_to_f16_kernel<<<dim3(H * H / 1024), dim3(256), 0, stream>>>(Wo, wo_h, H * H);

  // 2) projections: p{q,k,v} = X @ W^T + b   (f32 A converted in staging)
  {
    dim3 g(H / 128, MS / 128, 1), b(256);
    gemm128x128<float, false, true, _Float16><<<g, b, 0, stream>>>(Q,   wq_h, bq, pq, MS, H, H, 0, 0, 0);
    gemm128x128<float, false, true, _Float16><<<g, b, 0, stream>>>(Kin, wk_h, bk, pk, MS, H, H, 0, 0, 0);
    gemm128x128<float, false, true, _Float16><<<g, b, 0, stream>>>(V,   wv_h, bv, pv, MS, H, H, 0, 0, 0);
  }

  // 3) scores[b] = pq[b] @ pk[b]^T   (f32 out; scale/mask deferred to softmax)
  {
    dim3 g(SK / 128, SQ / 128, B), b(256);
    gemm128x128<_Float16, false, false, float><<<g, b, 0, stream>>>(
        pq, pk, nullptr, sc, SQ, SK, H,
        (long long)SQ * H, (long long)SK * H, (long long)SQ * SK);
  }

  // 4) P = softmax(mask ? sc/8 : -inf), per row
  softmax_mask_kernel<<<dim3(B * SQ), dim3(256), 0, stream>>>(sc, mask, P, SK);

  // 5) mh[b] = P[b] @ pv[b]   (B is [K,N] -> transposed staging)
  {
    dim3 g(H / 128, SQ / 128, B), b(256);
    gemm128x128<_Float16, true, false, _Float16><<<g, b, 0, stream>>>(
        P, pv, nullptr, mh, SQ, H, SK,
        (long long)SQ * SK, (long long)SK * H, (long long)SQ * H);
  }

  // 6) out = mh @ Wo^T + bo   (f32 out to d_out)
  {
    dim3 g(H / 128, MS / 128, 1), b(256);
    gemm128x128<_Float16, false, true, float><<<g, b, 0, stream>>>(
        mh, wo_h, bo, out, MS, H, H, 0, 0, 0);
  }
}
